// ShockCorrector_75617194213866
// MI455X (gfx1250) — compile-verified, matrix operand order checked by reading, not run on hardware
//
#include <hip/hip_runtime.h>
#include <hip/hip_bf16.h>
#include <math.h>

// Problem constants (from reference)
#define NXc 1024
#define NTc 128
#define Bc  4
#define BTc (Bc * NTc)            // 512
#define Hc  64
#define Lc  4
#define GHc 64
#define Ec  (2 * (NXc - 1))       // 2046
#define DXc 0.002f
#define DTc 0.01f
#define LN_EPSc 1e-5f

typedef __attribute__((ext_vector_type(16))) _Float16 v16h;
typedef __attribute__((ext_vector_type(8)))  _Float16 v8h;
typedef __attribute__((ext_vector_type(8)))  float    v8f;

__device__ __forceinline__ float gelu_f(float x) {
    // exact GELU: 0.5*x*(1+erf(x/sqrt(2)))
    return 0.5f * x * (1.0f + erff(x * 0.70710678118654752f));
}

// Load one 16x32 f16 fragment from an LDS row-major array (stride in halves).
// CDNA5 16-bit A-matrix layout (ISA 7.12.2): lanes 0-15 -> M=0..15 (low-K half
// base), lanes 16-31 -> same M, K-base +8; halves 0..7 = K kb+0..7,
// halves 8..15 = K kb+16..23. Two 16-byte LDS loads per lane.
__device__ __forceinline__ v16h ld_tile16(const _Float16* __restrict__ base,
                                          int row0, int k0, int stride, int lane) {
    int m  = lane & 15;
    int kb = (lane >> 4) << 3;
    const _Float16* p = base + (row0 + m) * stride + k0 + kb;
    v8h lo = *reinterpret_cast<const v8h*>(p);
    v8h hi = *reinterpret_cast<const v8h*>(p + 16);
    v16h r;
#pragma unroll
    for (int i = 0; i < 8; ++i) { r[i] = lo[i]; r[i + 8] = hi[i]; }
    return r;
}

__device__ __forceinline__ v8f wmma_f16f32(v16h a, v16h b, v8f c) {
    return __builtin_amdgcn_wmma_f32_16x16x32_f16(
        /*neg_a=*/false, a, /*neg_b=*/false, b,
        /*c_mod=*/(short)0, c, /*reuse_a=*/false, /*reuse_b=*/false);
}

// CDNA5 async copy: 16 bytes global -> LDS, tracked on ASYNCcnt (no VGPR data).
// lds_off is the byte address within this workgroup's LDS allocation.
__device__ __forceinline__ void async_g2l_b128(unsigned lds_off, const void* gaddr) {
    asm volatile("global_load_async_to_lds_b128 %0, %1, off"
                 :: "v"(lds_off), "v"(gaddr) : "memory");
}
__device__ __forceinline__ void wait_asynccnt0() {
    asm volatile("s_wait_asynccnt 0" ::: "memory");
}

// ---------------------------------------------------------------------------
// Kernel 1: derivative features + h0 = in_proj(u)
// ---------------------------------------------------------------------------
__global__ __launch_bounds__(256) void feat_h0_kernel(
    const float* __restrict__ u, const float* __restrict__ inw,
    const float* __restrict__ inb, _Float16* __restrict__ h0,
    float* __restrict__ feat) {
    int idx = blockIdx.x * blockDim.x + threadIdx.x;     // over BT*N nodes
    if (idx >= BTc * NXc) return;
    int n = idx % NXc;
    int bt = idx / NXc;
    int t = bt % NTc;
    float uc = u[idx];
    float dx;
    if (n == 0)           dx = (u[idx + 1] - uc) / DXc;
    else if (n == NXc-1)  dx = (uc - u[idx - 1]) / DXc;
    else                  dx = (u[idx + 1] - u[idx - 1]) / (2.0f * DXc);
    float dt;
    if (t == 0)           dt = (u[idx + NXc] - uc) / DTc;
    else if (t == NTc-1)  dt = (uc - u[idx - NXc]) / DTc;
    else                  dt = (u[idx + NXc] - u[idx - NXc]) / (2.0f * DTc);
    feat[(size_t)idx * 2 + 0] = log1pf(fabsf(dx));
    feat[(size_t)idx * 2 + 1] = log1pf(fabsf(dt));
    _Float16* hrow = h0 + (size_t)idx * Hc;
#pragma unroll
    for (int k = 0; k < Hc; ++k) hrow[k] = (_Float16)(uc * inw[k] + inb[k]);
}

// ---------------------------------------------------------------------------
// Kernel 2: all per-layer edge gates (depend only on edge features)
// ---------------------------------------------------------------------------
struct GateW {
    const float* g1w[Lc]; const float* g1b[Lc];
    const float* g2w[Lc]; const float* g2b[Lc];
};

__global__ __launch_bounds__(256) void gates_kernel(
    const float* __restrict__ u, const float* __restrict__ feat,
    GateW G, float* __restrict__ gates) {
    int idx = blockIdx.x * blockDim.x + threadIdx.x;     // over BT*E edges
    if (idx >= BTc * Ec) return;
    int e = idx % Ec;
    int bt = idx / Ec;
    int src, dst;
    if (e < NXc - 1) { src = e + 1; dst = e; }
    else { int j = e - (NXc - 1); src = j; dst = j + 1; }
    float f[5];
    f[0] = fabsf(u[(size_t)bt * NXc + dst] - u[(size_t)bt * NXc + src]);
    f[1] = fabsf(feat[((size_t)bt * NXc + dst) * 2 + 0]);
    f[2] = fabsf(feat[((size_t)bt * NXc + src) * 2 + 0]);
    f[3] = fabsf(feat[((size_t)bt * NXc + dst) * 2 + 1]);
    f[4] = fabsf(feat[((size_t)bt * NXc + src) * 2 + 1]);
    for (int l = 0; l < Lc; ++l) {
        const float* w1 = G.g1w[l];
        const float* b1 = G.g1b[l];
        const float* w2 = G.g2w[l];
        float acc = G.g2b[l][0];
        for (int j = 0; j < GHc; ++j) {
            float hsum = b1[j];
#pragma unroll
            for (int i = 0; i < 5; ++i) hsum += f[i] * w1[i * GHc + j];
            acc += gelu_f(hsum) * w2[j];
        }
        gates[(((size_t)l * BTc) + bt) * Ec + e] = 1.0f / (1.0f + expf(-acc));
    }
}

// ---------------------------------------------------------------------------
// Kernel 3: one GNN layer for a 64-node tile of one bt-slice (WMMA core)
// ---------------------------------------------------------------------------
struct LayerW {
    const float *m1w, *m1b, *m2w, *m2b;
    const float *u1w, *u1b, *u2w, *u2b;
    const float *lng, *lnb;
};

// dynamic LDS layout (halves then floats):
//   sA   : 128*160  msg_in (later reused as 64*128 upd_in)  -- at LDS offset 0
//   sH   : 128*64   hidden
//   sG   : 128*64   gated messages
//   sW1t : 64*160   msg1.w transposed (K padded 130->160)
//   sW2t : 64*64    msg2.w transposed
//   sU1t : 64*128   upd1.w transposed
//   sU2t : 64*64    upd2.w transposed
//   sB   : 384 f32  biases + ln params
//   sGate: 128 f32  per-local-edge gate
//   sOut : 64*64 f32 residual pre-LN
#define SMEM_HALVES (128*160 + 128*64 + 128*64 + 64*160 + 64*64 + 64*128 + 64*64)
#define SMEM_FLOATS (384 + 128 + 64*64)
#define SMEM_BYTES  ((size_t)SMEM_HALVES * 2 + (size_t)SMEM_FLOATS * 4)

__global__ __launch_bounds__(256) void layer_kernel(
    const _Float16* __restrict__ hIn, _Float16* __restrict__ hOut,
    const float* __restrict__ gatesL, LayerW W) {
    extern __shared__ char smemraw[];
    _Float16* sA    = (_Float16*)smemraw;      // at dynamic-LDS byte offset 0
    _Float16* sH    = sA + 128 * 160;
    _Float16* sG    = sH + 128 * 64;
    _Float16* sW1t  = sG + 128 * 64;
    _Float16* sW2t  = sW1t + 64 * 160;
    _Float16* sU1t  = sW2t + 64 * 64;
    _Float16* sU2t  = sU1t + 64 * 128;
    float*    sB    = (float*)(sU2t + 64 * 64);
    float*    sGate = sB + 384;
    float*    sOut  = sGate + 128;

    const int tid  = threadIdx.x;
    const int lane = tid & 31;
    const int wave = tid >> 5;
    const int bt   = blockIdx.y;
    const int n0   = blockIdx.x * 64;

    // ---- async-stage msg_in h blocks: global h rows are contiguous 128B ----
    // msg_in row le: cols 0..63 = h[dst], cols 64..127 = h[src].
    // 128 rows x 2 blocks x 8 chunks of 16B = 2048 lane-copies (8 per thread).
    {
        const char* hbase = (const char*)(hIn + (size_t)bt * NXc * Hc);
#pragma unroll
        for (int q = 0; q < 8; ++q) {
            int i = tid + 256 * q;          // 0..2047
            int le = i >> 4;
            int c  = i & 15;
            int half = c >> 3;              // 0 = dst block, 1 = src block
            int j  = c & 7;                 // 16B chunk within 64-half block
            int ln = le & 63, n = n0 + ln;
            bool right = le < 64;
            int srcn = right ? (n + 1 < NXc ? n + 1 : NXc - 1)
                             : (n - 1 >= 0 ? n - 1 : 0);
            int node = half ? srcn : n;
            unsigned lds = (unsigned)(le * 320 + half * 128 + j * 16);
            const char* g = hbase + (size_t)node * (Hc * 2) + j * 16;
            async_g2l_b128(lds, g);
        }
    }
    // tail columns of msg_in: edge_attr + zero padding (cols 128..159)
    for (int i = tid; i < 128 * 32; i += 256) {
        int le = i >> 5, k = 128 + (i & 31);
        bool right = le < 64;
        _Float16 v;
        if (k == 128)      v = (_Float16)(right ? -DXc : DXc);
        else if (k == 129) v = (_Float16)DXc;
        else               v = (_Float16)0.0f;
        sA[le * 160 + k] = v;
    }

    // ---- stage weights (transposed, f16) + biases + gates --------------
    for (int i = tid; i < 64 * 160; i += 256) {
        int n = i / 160, k = i % 160;
        sW1t[i] = (k < 2 * Hc + 2) ? (_Float16)W.m1w[k * Hc + n] : (_Float16)0.0f;
    }
    for (int i = tid; i < 64 * 64; i += 256) {
        int n = i >> 6, k = i & 63;
        sW2t[i] = (_Float16)W.m2w[k * Hc + n];
        sU2t[i] = (_Float16)W.u2w[k * Hc + n];
    }
    for (int i = tid; i < 64 * 128; i += 256) {
        int n = i >> 7, k = i & 127;
        sU1t[i] = (_Float16)W.u1w[k * Hc + n];
    }
    if (tid < 64) {
        sB[tid]       = W.m1b[tid];
        sB[64  + tid] = W.m2b[tid];
        sB[128 + tid] = W.u1b[tid];
        sB[192 + tid] = W.u2b[tid];
        sB[256 + tid] = W.lng[tid];
        sB[320 + tid] = W.lnb[tid];
    }
    if (tid < 128) {
        int le = tid, ln = le & 63, n = n0 + ln;
        bool right = le < 64;
        bool valid = right ? (n <= NXc - 2) : (n >= 1);
        int  e     = right ? n : (NXc - 1 + n - 1);
        float g = gatesL[(size_t)bt * Ec + e];   // index always in range
        sGate[le] = valid ? g : 0.0f;
    }
    wait_asynccnt0();
    __syncthreads();

    // ---- msg hidden = gelu(msg_in @ W1 + b1)   128x64 ------------------
    for (int t = wave; t < 32; t += 8) {
        int rt = t >> 2, ct = t & 3;
        v8f acc = {};
#pragma unroll
        for (int kc = 0; kc < 5; ++kc)
            acc = wmma_f16f32(ld_tile16(sA,   rt * 16, kc * 32, 160, lane),
                              ld_tile16(sW1t, ct * 16, kc * 32, 160, lane), acc);
        int n = ct * 16 + (lane & 15);
        float bias = sB[n];
        int mbase = rt * 16 + ((lane >> 4) << 3);
#pragma unroll
        for (int r = 0; r < 8; ++r)
            sH[(mbase + r) * 64 + n] = (_Float16)gelu_f(acc[r] + bias);
    }
    __syncthreads();

    // ---- gated msg = gate * (hidden @ W2 + b2)   128x64 ----------------
    for (int t = wave; t < 32; t += 8) {
        int rt = t >> 2, ct = t & 3;
        v8f acc = {};
#pragma unroll
        for (int kc = 0; kc < 2; ++kc)
            acc = wmma_f16f32(ld_tile16(sH,   rt * 16, kc * 32, 64, lane),
                              ld_tile16(sW2t, ct * 16, kc * 32, 64, lane), acc);
        int n = ct * 16 + (lane & 15);
        float bias = sB[64 + n];
        int mbase = rt * 16 + ((lane >> 4) << 3);
#pragma unroll
        for (int r = 0; r < 8; ++r) {
            int m = mbase + r;
            sG[m * 64 + n] = (_Float16)((acc[r] + bias) * sGate[m]);
        }
    }
    __syncthreads();

    // ---- upd_in = [h | agg],  agg[n] = gmsg_right(n) + gmsg_left(n) ----
    // cols 0..63 via async copy of contiguous global h rows
#pragma unroll
    for (int q = 0; q < 2; ++q) {
        int i = tid + 256 * q;              // 0..511
        int ln = i >> 3;
        int j  = i & 7;
        unsigned lds = (unsigned)(ln * 256 + j * 16);   // sA stride 128 halves
        const char* g = (const char*)(hIn + ((size_t)bt * NXc + n0 + ln) * Hc) + j * 16;
        async_g2l_b128(lds, g);
    }
    // cols 64..127 = aggregation of the two incident gated messages
    for (int i = tid; i < 64 * 64; i += 256) {
        int ln = i >> 6, kk = i & 63;
        sA[ln * 128 + 64 + kk] =
            (_Float16)((float)sG[ln * 64 + kk] + (float)sG[(64 + ln) * 64 + kk]);
    }
    wait_asynccnt0();
    __syncthreads();

    // ---- upd hidden = gelu(upd_in @ U1 + b)   64x64 --------------------
    for (int t = wave; t < 16; t += 8) {
        int rt = t >> 2, ct = t & 3;
        v8f acc = {};
#pragma unroll
        for (int kc = 0; kc < 4; ++kc)
            acc = wmma_f16f32(ld_tile16(sA,   rt * 16, kc * 32, 128, lane),
                              ld_tile16(sU1t, ct * 16, kc * 32, 128, lane), acc);
        int n = ct * 16 + (lane & 15);
        float bias = sB[128 + n];
        int mbase = rt * 16 + ((lane >> 4) << 3);
#pragma unroll
        for (int r = 0; r < 8; ++r)
            sH[(mbase + r) * 64 + n] = (_Float16)gelu_f(acc[r] + bias);
    }
    __syncthreads();

    // ---- residual = h + (hidden @ U2 + b)   64x64 f32 ------------------
    for (int t = wave; t < 16; t += 8) {
        int rt = t >> 2, ct = t & 3;
        v8f acc = {};
#pragma unroll
        for (int kc = 0; kc < 2; ++kc)
            acc = wmma_f16f32(ld_tile16(sH,   rt * 16, kc * 32, 64, lane),
                              ld_tile16(sU2t, ct * 16, kc * 32, 64, lane), acc);
        int n = ct * 16 + (lane & 15);
        float bias = sB[192 + n];
        int mbase = rt * 16 + ((lane >> 4) << 3);
#pragma unroll
        for (int r = 0; r < 8; ++r) {
            int m = mbase + r;
            sOut[m * 64 + n] = acc[r] + bias + (float)sA[m * 128 + n];
        }
    }
    __syncthreads();

    // ---- LayerNorm + writeout ------------------------------------------
    if (tid < 64) {
        const float* row = sOut + tid * 64;
        float mean = 0.0f;
        for (int k = 0; k < 64; ++k) mean += row[k];
        mean *= (1.0f / 64.0f);
        float var = 0.0f;
        for (int k = 0; k < 64; ++k) { float d = row[k] - mean; var += d * d; }
        var *= (1.0f / 64.0f);
        float is = rsqrtf(var + LN_EPSc);
        _Float16* orow = hOut + ((size_t)bt * NXc + n0 + tid) * Hc;
        for (int k = 0; k < 64; ++k)
            orow[k] = (_Float16)((row[k] - mean) * is * sB[256 + k] + sB[320 + k]);
    }
}

// ---------------------------------------------------------------------------
// Kernel 4: delta_u = h @ w_out + b_out
// ---------------------------------------------------------------------------
__global__ __launch_bounds__(256) void out_kernel(
    const _Float16* __restrict__ h, const float* __restrict__ ow,
    const float* __restrict__ ob, float* __restrict__ out) {
    int idx = blockIdx.x * blockDim.x + threadIdx.x;
    if (idx >= BTc * NXc) return;
    const _Float16* row = h + (size_t)idx * Hc;
    float acc = ob[0];
#pragma unroll
    for (int k = 0; k < Hc; ++k) acc += (float)row[k] * ow[k];
    out[idx] = acc;
}

// ---------------------------------------------------------------------------
extern "C" void kernel_launch(void* const* d_in, const int* in_sizes, int n_in,
                              void* d_out, int out_size, void* d_ws, size_t ws_size,
                              hipStream_t stream) {
    (void)out_size; (void)ws_size;
    const float *u = nullptr, *inw = nullptr, *inb = nullptr;
    const float *ow = nullptr, *ob = nullptr;
    LayerW LW[Lc];
    GateW  GW;

    if (n_in >= 65 && in_sizes[0] == Bc * NTc * NXc * 2) {
        // insertion-order flatten: x,u, in_proj(w,b), out(w,b),
        // layers[l]: msg1(w,b) msg2(w,b) gate1(w,b) gate2(w,b) upd1(w,b) upd2(w,b) ln_g ln_b,
        // edge_src, edge_dst, edge_attr
        u   = (const float*)d_in[1];
        inw = (const float*)d_in[2];  inb = (const float*)d_in[3];
        ow  = (const float*)d_in[4];  ob  = (const float*)d_in[5];
        for (int l = 0; l < Lc; ++l) {
            int b0 = 6 + 14 * l;
            LW[l].m1w = (const float*)d_in[b0 + 0];  LW[l].m1b = (const float*)d_in[b0 + 1];
            LW[l].m2w = (const float*)d_in[b0 + 2];  LW[l].m2b = (const float*)d_in[b0 + 3];
            GW.g1w[l] = (const float*)d_in[b0 + 4];  GW.g1b[l] = (const float*)d_in[b0 + 5];
            GW.g2w[l] = (const float*)d_in[b0 + 6];  GW.g2b[l] = (const float*)d_in[b0 + 7];
            LW[l].u1w = (const float*)d_in[b0 + 8];  LW[l].u1b = (const float*)d_in[b0 + 9];
            LW[l].u2w = (const float*)d_in[b0 + 10]; LW[l].u2b = (const float*)d_in[b0 + 11];
            LW[l].lng = (const float*)d_in[b0 + 12]; LW[l].lnb = (const float*)d_in[b0 + 13];
        }
    } else {
        // JAX sorted-leaf flatten: edge_attr, edge_dst, edge_src, in_proj.b, in_proj.w,
        // layers[l]: gate1.b,gate1.w,gate2.b,gate2.w,ln_b,ln_g,msg1.b,msg1.w,msg2.b,msg2.w,
        //            upd1.b,upd1.w,upd2.b,upd2.w, then out.b, out.w, u, x
        inb = (const float*)d_in[3];  inw = (const float*)d_in[4];
        for (int l = 0; l < Lc; ++l) {
            int b0 = 5 + 14 * l;
            GW.g1b[l] = (const float*)d_in[b0 + 0];  GW.g1w[l] = (const float*)d_in[b0 + 1];
            GW.g2b[l] = (const float*)d_in[b0 + 2];  GW.g2w[l] = (const float*)d_in[b0 + 3];
            LW[l].lnb = (const float*)d_in[b0 + 4];  LW[l].lng = (const float*)d_in[b0 + 5];
            LW[l].m1b = (const float*)d_in[b0 + 6];  LW[l].m1w = (const float*)d_in[b0 + 7];
            LW[l].m2b = (const float*)d_in[b0 + 8];  LW[l].m2w = (const float*)d_in[b0 + 9];
            LW[l].u1b = (const float*)d_in[b0 + 10]; LW[l].u1w = (const float*)d_in[b0 + 11];
            LW[l].u2b = (const float*)d_in[b0 + 12]; LW[l].u2w = (const float*)d_in[b0 + 13];
        }
        ob = (const float*)d_in[61]; ow = (const float*)d_in[62];
        u  = (const float*)d_in[63];
    }

    // workspace carve: hA, hB (f16 BT*N*H each), gates (f32 L*BT*E), feat (f32 BT*N*2)
    const size_t nH = (size_t)BTc * NXc * Hc;
    _Float16* hA    = (_Float16*)d_ws;
    _Float16* hB    = hA + nH;
    float*    gates = (float*)(hB + nH);
    float*    feat  = gates + (size_t)Lc * BTc * Ec;

    feat_h0_kernel<<<(BTc * NXc + 255) / 256, 256, 0, stream>>>(u, inw, inb, hA, feat);
    gates_kernel<<<(BTc * Ec + 255) / 256, 256, 0, stream>>>(u, feat, GW, gates);

    // allow >64KB dynamic LDS (no-op / ignored if unnecessary on this stack)
    (void)hipFuncSetAttribute(reinterpret_cast<const void*>(layer_kernel),
                              hipFuncAttributeMaxDynamicSharedMemorySize,
                              (int)SMEM_BYTES);

    _Float16* hcur = hA;
    _Float16* hnxt = hB;
    for (int l = 0; l < Lc; ++l) {
        layer_kernel<<<dim3(NXc / 64, BTc), 256, SMEM_BYTES, stream>>>(
            hcur, hnxt, gates + (size_t)l * BTc * Ec, LW[l]);
        _Float16* tmp = hcur; hcur = hnxt; hnxt = tmp;
    }
    out_kernel<<<(BTc * NXc + 255) / 256, 256, 0, stream>>>(hcur, ow, ob, (float*)d_out);
}
